// MainModule_57501022158924
// MI455X (gfx1250) — matrix operand order repaired and checked by reading, not compiled, not run on hardware
//
#include <hip/hip_runtime.h>
#include <math.h>

#define BB 8
#define TT 4096
#define BT (BB*TT)
#define DD 64
#define NLAY 11
#define NHD 8

typedef __attribute__((ext_vector_type(16))) __bf16 v16bf;
typedef __attribute__((ext_vector_type(8)))  __bf16 v8bf;
typedef __attribute__((ext_vector_type(8)))  float  v8f;

#define WMMA_BF16(A, Bf, C) \
    __builtin_amdgcn_wmma_f32_16x16x32_bf16(false, (A), false, (Bf), (short)0, (C), false, false)

// ---------- fragment helpers ----------
// A and B fragments are stored pre-swizzled so each lane's 16 bf16 = 32 contiguous bytes.
// A-frag LDS layout: hf[(wave*2 + kb)*32 + lane][16]
__device__ __forceinline__ v16bf load_a_frag(const __bf16* hf, int wave, int kb, int lane) {
    return *(const v16bf*)(hf + (size_t)(((wave * 2 + kb) * 32) + lane) * 16);
}
// B-frag global layout: wf[(kb*4 + nt)*32 + lane][16]
__device__ __forceinline__ v16bf load_b_frag(const __bf16* wf, int kb, int nt, int lane) {
    return *(const v16bf*)(wf + (size_t)(((kb * 4 + nt) * 32) + lane) * 16);
}

// ---------- weight convert into B-fragment order ----------
// out[m][L][kb][nt][lane][e]  with  B[k][n] = W[n][k]
__global__ void k_wcvt(const float* __restrict__ Wq, const float* __restrict__ Wk,
                       const float* __restrict__ Wv, const float* __restrict__ W1,
                       const float* __restrict__ W2, __bf16* __restrict__ out) {
    int i = blockIdx.x * blockDim.x + threadIdx.x;
    if (i >= 5 * NLAY * 4096) return;
    int m = i / (NLAY * 4096);
    int r = i % (NLAY * 4096);
    int L = r / 4096;
    int e4 = r % 4096;
    int kb = e4 >> 11, nt = (e4 >> 9) & 3, lane = (e4 >> 4) & 31, e = e4 & 15;
    int vv = e >> 1, p = e & 1;
    int k = kb * 32 + ((lane >> 4) << 3) + ((2 * vv + p) & 7) + (((2 * vv + p) >> 3) << 4);
    int n = nt * 16 + (lane & 15);
    const float* src = (m == 0) ? Wq : (m == 1) ? Wk : (m == 2) ? Wv : (m == 3) ? W1 : W2;
    out[i] = (__bf16)src[L * 4096 + n * 64 + k];
}

// ---------- LN -> A-fragment-swizzled bf16 LDS (ds_store_b128) ----------
// Per lane-slot of 16 values: two contiguous K-runs of 8 -> two v8bf stores.
__device__ __forceinline__ void ln_to_frag(const float* __restrict__ x, int t0, int tid,
                                           const float* __restrict__ g,
                                           const float* __restrict__ be,
                                           __bf16* hf) {
    if (tid < 64) {
        const float4* xr4 = (const float4*)(x + (size_t)(t0 + tid) * DD);
        float lnv[64];
        float s = 0.f, s2 = 0.f;
#pragma unroll
        for (int i = 0; i < 16; ++i) {
            float4 xv = xr4[i];
            lnv[4 * i + 0] = xv.x; lnv[4 * i + 1] = xv.y;
            lnv[4 * i + 2] = xv.z; lnv[4 * i + 3] = xv.w;
            s  += xv.x + xv.y + xv.z + xv.w;
            s2 += xv.x * xv.x + xv.y * xv.y + xv.z * xv.z + xv.w * xv.w;
        }
        float mean = s * (1.f / DD);
        float var  = s2 * (1.f / DD) - mean * mean;
        float inv  = rsqrtf(var + 1e-5f);
        int w = tid >> 4, r = tid & 15;
#pragma unroll
        for (int kb = 0; kb < 2; ++kb)
#pragma unroll
            for (int half = 0; half < 2; ++half) {
                __bf16* dst = hf + (size_t)(((w * 2 + kb) * 32) + half * 16 + r) * 16;
                v8bf p0, p1;
#pragma unroll
                for (int j = 0; j < 8; ++j) {
                    int k0 = kb * 32 + half * 8 + j;
                    int k1 = kb * 32 + 16 + half * 8 + j;
                    p0[j] = (__bf16)((lnv[k0] - mean) * inv * g[k0] + be[k0]);
                    p1[j] = (__bf16)((lnv[k1] - mean) * inv * g[k1] + be[k1]);
                }
                *(v8bf*)dst = p0;
                *(v8bf*)(dst + 8) = p1;
            }
    }
}

// ---------- conv frontend ----------

__global__ void k_conv1(const float* __restrict__ spec, const float* __restrict__ w,
                        const float* __restrict__ bias, float* __restrict__ x1) {
    int bt = blockIdx.x * blockDim.x + threadIdx.x;
    if (bt >= BT) return;
    int b = bt >> 12, t = bt & (TT - 1);
    const float* sp = spec + (size_t)b * 81 * TT;
    int nsub = TT - t; if (nsub > 3) nsub = 3;
    float bsum[8];
#pragma unroll
    for (int c = 0; c < 8; ++c) {
        float s = 0.f;
        for (int i = 0; i < 3; ++i) if (i < nsub) s += bias[i * 8 + c];
        bsum[c] = s;
    }
    float wl[72];
#pragma unroll
    for (int i = 0; i < 72; ++i) wl[i] = w[i];   // w[(i*8+c)*3+kw]
    for (int f2 = 0; f2 < 26; ++f2) {
        float s[5][3];
#pragma unroll
        for (int ff = 0; ff < 5; ++ff)
#pragma unroll
            for (int i = 0; i < 3; ++i)
                s[ff][i] = (i < nsub) ? sp[(size_t)(3 * f2 + ff) * TT + t + i] : 0.f;
        float mx[8];
#pragma unroll
        for (int c = 0; c < 8; ++c) mx[c] = -3.4e38f;
#pragma unroll
        for (int fp = 0; fp < 3; ++fp)
#pragma unroll
            for (int c = 0; c < 8; ++c) {
                float acc = bsum[c];
#pragma unroll
                for (int i = 0; i < 3; ++i)
#pragma unroll
                    for (int kw = 0; kw < 3; ++kw)
                        acc += wl[(i * 8 + c) * 3 + kw] * s[fp + kw][i];
                mx[c] = fmaxf(mx[c], acc);
            }
#pragma unroll
        for (int c = 0; c < 8; ++c)
            x1[(((size_t)(b * 8 + c)) * TT + t) * 26 + f2] = fmaxf(mx[c], 0.f);
    }
}

__global__ __launch_bounds__(256) void k_conv2(const float* __restrict__ x1,
                                               const float* __restrict__ w,
                                               const float* __restrict__ bias,
                                               float* __restrict__ x2) {
    __shared__ float in[8 * 208];   // 8 tokens x 8cin x 26f
    int t0 = blockIdx.x * 8;
    for (int i = threadIdx.x; i < 8 * 208; i += 256) {
        int tok = i / 208, r = i % 208, cin = r / 26, f = r % 26;
        int bt = t0 + tok, b = bt >> 12, t = bt & (TT - 1);
        in[i] = x1[(((size_t)(b * 8 + cin)) * TT + t) * 26 + f];
    }
    __syncthreads();
    int tok = threadIdx.x >> 5, c = threadIdx.x & 31;
    const float* ip = in + tok * 208;
    float wl[96];
#pragma unroll
    for (int i = 0; i < 96; ++i) wl[i] = w[c * 96 + i];   // w[c][cin][kw]
    int bt = t0 + tok, b = bt >> 12, t = bt & (TT - 1);
    float bc = bias[c];
    for (int f2 = 0; f2 < 5; ++f2) {
        float mx = -3.4e38f;
#pragma unroll
        for (int fp = 0; fp < 3; ++fp) {
            float acc = bc;
            int fb = 3 * f2 + fp;
#pragma unroll
            for (int cin = 0; cin < 8; ++cin)
#pragma unroll
                for (int kw = 0; kw < 12; ++kw)
                    acc += wl[cin * 12 + kw] * ip[cin * 26 + fb + kw];
            mx = fmaxf(mx, acc);
        }
        x2[(((size_t)(b * 32 + c)) * TT + t) * 5 + f2] = fmaxf(mx, 0.f);
    }
}

__global__ __launch_bounds__(256) void k_conv3(const float* __restrict__ x2,
                                               const float* __restrict__ w,
                                               const float* __restrict__ bias,
                                               float* __restrict__ x) {
    __shared__ float in[4 * 480];   // 4 tokens x 32cin x 3ti x 5f
    int t0 = blockIdx.x * 4;
    for (int i = threadIdx.x; i < 4 * 480; i += 256) {
        int tok = i / 480, r = i % 480, cin = r / 15, rr = r % 15, ti = rr / 5, f = rr % 5;
        int bt = t0 + tok, b = bt >> 12, t = bt & (TT - 1);
        in[i] = (t + ti < TT) ? x2[(((size_t)(b * 32 + cin)) * TT + (t + ti)) * 5 + f] : 0.f;
    }
    __syncthreads();
    int tok = threadIdx.x >> 6, c = threadIdx.x & 63;
    int bt = t0 + tok, t = bt & (TT - 1);
    int nsub = TT - t; if (nsub > 3) nsub = 3;
    float bsum = 0.f;
    for (int i = 0; i < 3; ++i) if (i < nsub) bsum += bias[i * 64 + c];
    const float* ip = in + tok * 480;
    float mx = -3.4e38f;
#pragma unroll
    for (int f = 0; f < 3; ++f) {
        float acc = bsum;
#pragma unroll
        for (int i = 0; i < 3; ++i)
            for (int cin = 0; cin < 32; ++cin) {
                const float* wp = w + (((i * 64 + c) * 32 + cin) * 3);
                const float* xp = ip + cin * 15 + i * 5 + f;
                acc += wp[0] * xp[0] + wp[1] * xp[1] + wp[2] * xp[2];
            }
        mx = fmaxf(mx, acc);
    }
    x[(size_t)bt * DD + c] = fmaxf(mx, 0.f);
}

// ---------- transformer ----------

// LN(x) -> A-frags in LDS -> q/k/v = h @ Wt + b via v_wmma_f32_16x16x32_bf16
__global__ __launch_bounds__(128) void k_qkv(const float* __restrict__ x,
                                             const float* __restrict__ g, const float* __restrict__ be,
                                             const __bf16* __restrict__ wtq, const __bf16* __restrict__ wtk,
                                             const __bf16* __restrict__ wtv,
                                             const float* __restrict__ bq, const float* __restrict__ bk,
                                             const float* __restrict__ bv,
                                             float* __restrict__ q, float* __restrict__ k, float* __restrict__ v) {
    __shared__ __bf16 hf[4096];
    int tid = threadIdx.x;
    int t0 = blockIdx.x * 64;
    ln_to_frag(x, t0, tid, g, be, hf);
    __syncthreads();
    int wave = tid >> 5, lane = tid & 31;
    int half = lane >> 4, nl = lane & 15;
    v16bf a0 = load_a_frag(hf, wave, 0, lane);
    v16bf a1 = load_a_frag(hf, wave, 1, lane);

    const __bf16* wts[3]  = {wtq, wtk, wtv};
    const float*  bs[3]   = {bq, bk, bv};
    float*        outs[3] = {q, k, v};
#pragma unroll
    for (int m = 0; m < 3; ++m) {
#pragma unroll
        for (int nt = 0; nt < 4; ++nt) {
            v16bf b0 = load_b_frag(wts[m], 0, nt, lane);
            v16bf b1 = load_b_frag(wts[m], 1, nt, lane);
            v8f acc = {};
            acc = WMMA_BF16(a0, b0, acc);
            acc = WMMA_BF16(a1, b1, acc);
            int n = nt * 16 + nl;
            float bvv = bs[m][n];
#pragma unroll
            for (int vv = 0; vv < 8; ++vv) {
                int tt = t0 + wave * 16 + vv + 8 * half;
                outs[m][(size_t)tt * DD + n] = acc[vv] + bvv;
            }
        }
    }
}

// 6-key dilated local attention + residual into x (float4-vectorized)
__global__ void k_attn(const float* __restrict__ q, const float* __restrict__ k,
                       const float* __restrict__ v, const float* __restrict__ Er,
                       float* __restrict__ x, int layer) {
    int gid = blockIdx.x * blockDim.x + threadIdx.x;
    if (gid >= BT * NHD) return;
    int hh = gid & 7, bt = gid >> 3;
    int b = bt >> 12, t = bt & (TT - 1);
    const float4* qp4 = (const float4*)(q + (size_t)bt * DD + hh * 8);
    float4 q0 = qp4[0], q1 = qp4[1];
    float qv[8] = {q0.x, q0.y, q0.z, q0.w, q1.x, q1.y, q1.z, q1.w};
    // Er block for this head: 48 contiguous floats = 12 float4
    float er[48];
    {
        const float4* ep4 = (const float4*)(Er + hh * 48);
#pragma unroll
        for (int i = 0; i < 12; ++i) {
            float4 e = ep4[i];
            er[4 * i + 0] = e.x; er[4 * i + 1] = e.y; er[4 * i + 2] = e.z; er[4 * i + 3] = e.w;
        }
    }
    int step = 1 << layer;
    int sh[6] = {0, 0, step, 2 * step, 3 * step, 4 * step};
    float lg[6];
    float mxl = -3.4e38f;
#pragma unroll
    for (int j = 0; j < 6; ++j) {
        int ts = t - sh[j];
        float dot = 0.f;
        if (ts >= 0) {
            const float4* kp4 = (const float4*)(k + ((size_t)(b * TT + ts)) * DD + hh * 8);
            float4 k0 = kp4[0], k1 = kp4[1];
            dot = q0.x * k0.x + q0.y * k0.y + q0.z * k0.z + q0.w * k0.w +
                  q1.x * k1.x + q1.y * k1.y + q1.z * k1.z + q1.w * k1.w;
        }
        float qer = 0.f;
#pragma unroll
        for (int d = 0; d < 8; ++d) qer += qv[d] * er[d * 6 + j];
        float l = (dot + qer) * 0.35355339059327373f + ((dot == 0.f) ? 1e-9f : 0.f);
        lg[j] = l;
        mxl = fmaxf(mxl, l);
    }
    float sum = 0.f;
#pragma unroll
    for (int j = 0; j < 6; ++j) { lg[j] = expf(lg[j] - mxl); sum += lg[j]; }
    float inv = 1.f / sum;
    float4 o0 = {0.f, 0.f, 0.f, 0.f}, o1 = {0.f, 0.f, 0.f, 0.f};
#pragma unroll
    for (int j = 0; j < 6; ++j) {
        int ts = t - sh[j];
        if (ts >= 0) {
            const float4* vp4 = (const float4*)(v + ((size_t)(b * TT + ts)) * DD + hh * 8);
            float4 v0 = vp4[0], v1 = vp4[1];
            float a = lg[j] * inv;
            o0.x += a * v0.x; o0.y += a * v0.y; o0.z += a * v0.z; o0.w += a * v0.w;
            o1.x += a * v1.x; o1.y += a * v1.y; o1.z += a * v1.z; o1.w += a * v1.w;
        }
    }
    float4* xp4 = (float4*)(x + (size_t)bt * DD + hh * 8);
    float4 x0 = xp4[0], x1v = xp4[1];
    x0.x += o0.x; x0.y += o0.y; x0.z += o0.z; x0.w += o0.w;
    x1v.x += o1.x; x1v.y += o1.y; x1v.z += o1.z; x1v.w += o1.w;
    xp4[0] = x0; xp4[1] = x1v;
}

// FFN: LN2(x) -> gelu(h@W1t+b1)@W2t + b2, residual into x. Both GEMMs via WMMA.
__global__ __launch_bounds__(128) void k_ffn(float* __restrict__ x,
                                             const float* __restrict__ g, const float* __restrict__ be,
                                             const __bf16* __restrict__ wt1, const float* __restrict__ b1,
                                             const __bf16* __restrict__ wt2, const float* __restrict__ b2) {
    __shared__ __bf16 hf[4096];
    __shared__ __bf16 h2[4096];
    int tid = threadIdx.x;
    int t0 = blockIdx.x * 64;
    ln_to_frag(x, t0, tid, g, be, hf);
    __syncthreads();
    int wave = tid >> 5, lane = tid & 31;
    int half = lane >> 4, nl = lane & 15;
    v16bf a0 = load_a_frag(hf, wave, 0, lane);
    v16bf a1 = load_a_frag(hf, wave, 1, lane);
#pragma unroll
    for (int nt = 0; nt < 4; ++nt) {
        v16bf b0 = load_b_frag(wt1, 0, nt, lane);
        v16bf b1f = load_b_frag(wt1, 1, nt, lane);
        v8f acc = {};
        acc = WMMA_BF16(a0, b0, acc);
        acc = WMMA_BF16(a1, b1f, acc);
        int n = nt * 16 + nl;
        // scatter GELU result directly into A-fragment layout of h2
        int kb = n >> 5;
        int kr = n & 31;
        int half2 = (kr >> 3) & 1;
        int e = (kr & 7) + ((kr & 16) ? 8 : 0);
        float bvv = b1[n];
#pragma unroll
        for (int vv = 0; vv < 8; ++vv) {
            int r = vv + 8 * half;  // row within the wave's 16-token tile
            float u = acc[vv] + bvv;
            float gel = 0.5f * u * (1.f + erff(u * 0.70710678118f));   // exact GELU
            h2[(size_t)(((wave * 2 + kb) * 32) + half2 * 16 + r) * 16 + e] = (__bf16)gel;
        }
    }
    __syncthreads();
    v16bf c0 = load_a_frag(h2, wave, 0, lane);
    v16bf c1 = load_a_frag(h2, wave, 1, lane);
#pragma unroll
    for (int nt = 0; nt < 4; ++nt) {
        v16bf b0 = load_b_frag(wt2, 0, nt, lane);
        v16bf b1f = load_b_frag(wt2, 1, nt, lane);
        v8f acc = {};
        acc = WMMA_BF16(c0, b0, acc);
        acc = WMMA_BF16(c1, b1f, acc);
        int n = nt * 16 + nl;
        float bvv = b2[n];
#pragma unroll
        for (int vv = 0; vv < 8; ++vv) {
            int tt = t0 + wave * 16 + vv + 8 * half;
            x[(size_t)tt * DD + n] += acc[vv] + bvv;
        }
    }
}

// output head: sigmoid(x @ out_w.T + out_b) -> (B,2,T)
__global__ void k_out(const float* __restrict__ x, const float* __restrict__ ow,
                      const float* __restrict__ ob, float* __restrict__ out) {
    int bt = blockIdx.x * blockDim.x + threadIdx.x;
    if (bt >= BT) return;
    int b = bt >> 12, t = bt & (TT - 1);
    const float4* xr4 = (const float4*)(x + (size_t)bt * DD);
    float a0 = ob[0], a1 = ob[1];
#pragma unroll
    for (int i = 0; i < 16; ++i) {
        float4 xv = xr4[i];
        a0 += xv.x * ow[4 * i] + xv.y * ow[4 * i + 1] + xv.z * ow[4 * i + 2] + xv.w * ow[4 * i + 3];
        a1 += xv.x * ow[64 + 4 * i] + xv.y * ow[64 + 4 * i + 1] +
              xv.z * ow[64 + 4 * i + 2] + xv.w * ow[64 + 4 * i + 3];
    }
    out[((size_t)b * 2 + 0) * TT + t] = 1.f / (1.f + expf(-a0));
    out[((size_t)b * 2 + 1) * TT + t] = 1.f / (1.f + expf(-a1));
}

extern "C" void kernel_launch(void* const* d_in, const int* in_sizes, int n_in,
                              void* d_out, int out_size, void* d_ws, size_t ws_size,
                              hipStream_t stream) {
    const float* spec  = (const float*)d_in[0];
    const float* c1_w  = (const float*)d_in[1];
    const float* c1_b  = (const float*)d_in[2];
    const float* c2_w  = (const float*)d_in[3];
    const float* c2_b  = (const float*)d_in[4];
    const float* c3_w  = (const float*)d_in[5];
    const float* c3_b  = (const float*)d_in[6];
    const float* Wq    = (const float*)d_in[7];
    const float* bq    = (const float*)d_in[8];
    const float* Wk    = (const float*)d_in[9];
    const float* bk    = (const float*)d_in[10];
    const float* Wv    = (const float*)d_in[11];
    const float* bv    = (const float*)d_in[12];
    const float* Er    = (const float*)d_in[13];
    const float* g1    = (const float*)d_in[14];
    const float* be1   = (const float*)d_in[15];
    const float* W1    = (const float*)d_in[16];
    const float* b1    = (const float*)d_in[17];
    const float* W2    = (const float*)d_in[18];
    const float* b2    = (const float*)d_in[19];
    const float* g2    = (const float*)d_in[20];
    const float* be2   = (const float*)d_in[21];
    const float* out_w = (const float*)d_in[22];
    const float* out_b = (const float*)d_in[23];

    char* ws = (char*)d_ws;
    const size_t OFF_X = 512ull * 1024;               // x: BT*64*4 = 8 MB
    const size_t OFF_A = OFF_X + 9ull * 1024 * 1024;  // shared region
    __bf16* Wt = (__bf16*)ws;                         // 5*11*4096 bf16 (fragment-swizzled)
    float* x   = (float*)(ws + OFF_X);
    // frontend intermediates (dead before q/k/v are written):
    float* x1  = (float*)(ws + OFF_A);                       // 27 MB
    float* x2  = (float*)(ws + OFF_A + 28ull * 1024 * 1024); // 21 MB
    // transformer scratch (alias x1 region):
    float* q   = (float*)(ws + OFF_A);
    float* kk  = (float*)(ws + OFF_A + 9ull * 1024 * 1024);
    float* vv  = (float*)(ws + OFF_A + 18ull * 1024 * 1024);

    k_wcvt<<<(5 * NLAY * 4096 + 255) / 256, 256, 0, stream>>>(Wq, Wk, Wv, W1, W2, Wt);
    k_conv1<<<BT / 256, 256, 0, stream>>>(spec, c1_w, c1_b, x1);
    k_conv2<<<BT / 8, 256, 0, stream>>>(x1, c2_w, c2_b, x2);
    k_conv3<<<BT / 4, 256, 0, stream>>>(x2, c3_w, c3_b, x);

    for (int L = 0; L < NLAY; ++L) {
        const __bf16* wtq = Wt + (size_t)(0 * NLAY + L) * 4096;
        const __bf16* wtk = Wt + (size_t)(1 * NLAY + L) * 4096;
        const __bf16* wtv = Wt + (size_t)(2 * NLAY + L) * 4096;
        const __bf16* wt1 = Wt + (size_t)(3 * NLAY + L) * 4096;
        const __bf16* wt2 = Wt + (size_t)(4 * NLAY + L) * 4096;
        k_qkv<<<BT / 64, 128, 0, stream>>>(x, g1 + L * 64, be1 + L * 64,
                                           wtq, wtk, wtv,
                                           bq + L * 64, bk + L * 64, bv + L * 64,
                                           q, kk, vv);
        k_attn<<<(BT * NHD) / 256, 256, 0, stream>>>(q, kk, vv, Er + L * 8 * 8 * 6, x, L);
        k_ffn<<<BT / 64, 128, 0, stream>>>(x, g2 + L * 64, be2 + L * 64,
                                           wt1, b1 + L * 64, wt2, b2 + L * 64);
    }
    k_out<<<BT / 256, 256, 0, stream>>>(x, out_w, out_b, (float*)d_out);
}
